// Attention_64055142252858
// MI455X (gfx1250) — matrix operand order repaired and checked by reading, not compiled, not run on hardware
//
#include <hip/hip_runtime.h>
#include <hip/hip_bf16.h>

// ---------------------------------------------------------------------------
// MI455X (gfx1250, wave32) multi-head attention, bf16 WMMA pipeline.
//   tok = transpose(x); Q/K/V = tok @ W^T + b (async-staged WMMA GEMM);
//   flash attention per (batch, head, 128-row tile); out = AO @ Wp^T + bp.
// All matmuls: v_wmma_f32_16x16x32_bf16 (f32 accumulate).
// Tile staging: GLOBAL_LOAD_ASYNC_TO_LDS_B128 (ASYNCcnt), double-buffered,
// in both the GEMM (A & W tiles) and attention (K tiles).
// ---------------------------------------------------------------------------

typedef __bf16 bf16t;
typedef __attribute__((ext_vector_type(16))) __bf16 v16bf;
typedef __attribute__((ext_vector_type(8)))  float  v8f;
typedef int v4i_vs __attribute__((vector_size(16)));   // matches builtin's V4i

#define C_DIM 512
#define S_DIM 4096
#define NH    8
#define HD    64
#define BATCH 2

#if defined(__has_builtin)
#  if __has_builtin(__builtin_amdgcn_global_load_async_to_lds_b128)
#    define HAS_ASYNC_LDS 1
#  endif
#endif
#ifndef HAS_ASYNC_LDS
#  define HAS_ASYNC_LDS 0
#endif

static __device__ __forceinline__ v8f v8f_zero() {
    v8f z = {0.f, 0.f, 0.f, 0.f, 0.f, 0.f, 0.f, 0.f};
    return z;
}

// Load a 16-element bf16 fragment as two 16-byte chunks (works for global & LDS).
static __device__ __forceinline__ v16bf ld_frag16(const bf16t* p0, const bf16t* p1) {
    union { v16bf v; uint4 q[2]; } u;
    u.q[0] = *reinterpret_cast<const uint4*>(p0);
    u.q[1] = *reinterpret_cast<const uint4*>(p1);
    return u.v;
}

// 16-byte global -> LDS copy: async DMA (ASYNCcnt) if available, else sync.
static __device__ __forceinline__ void copy16_g2l(bf16t* lds, const bf16t* g) {
#if HAS_ASYNC_LDS
    __builtin_amdgcn_global_load_async_to_lds_b128(
        (__attribute__((address_space(1))) v4i_vs*)g,
        (__attribute__((address_space(3))) v4i_vs*)lds, 0, 0);
#else
    *reinterpret_cast<uint4*>(lds) = *reinterpret_cast<const uint4*>(g);
#endif
}

static __device__ __forceinline__ void wait_async0() {
#if HAS_ASYNC_LDS
#  if __has_builtin(__builtin_amdgcn_s_wait_asynccnt)
    __builtin_amdgcn_s_wait_asynccnt(0);
#  else
    asm volatile("s_wait_asynccnt 0x0" ::: "memory");
#  endif
#endif
}

// ---------------------------------------------------------------------------
// Convert the four weight matrices fp32 -> bf16.
// ---------------------------------------------------------------------------
__global__ void cvt_w(const float* __restrict__ q, const float* __restrict__ k,
                      const float* __restrict__ v, const float* __restrict__ p,
                      bf16t* __restrict__ oq, bf16t* __restrict__ ok,
                      bf16t* __restrict__ ov, bf16t* __restrict__ op) {
    int i = blockIdx.x * blockDim.x + threadIdx.x;
    if (i < C_DIM * C_DIM) {
        oq[i] = (bf16t)q[i];
        ok[i] = (bf16t)k[i];
        ov[i] = (bf16t)v[i];
        op[i] = (bf16t)p[i];
    }
}

// ---------------------------------------------------------------------------
// x [B, C, S] fp32  ->  tok [B, S, C] bf16   (LDS 32x32 tile transpose)
// ---------------------------------------------------------------------------
__global__ void __launch_bounds__(256)
tok_transpose(const float* __restrict__ x, bf16t* __restrict__ tok) {
    const int b  = blockIdx.z;
    const int s0 = blockIdx.x * 32;
    const int c0 = blockIdx.y * 32;
    __shared__ float tile[32][33];
    const int tx = threadIdx.x, ty = threadIdx.y;   // 32 x 8
    const float* xb = x + (size_t)b * C_DIM * S_DIM;
    bf16t* tb = tok + (size_t)b * S_DIM * C_DIM;
#pragma unroll
    for (int i = 0; i < 4; ++i)
        tile[ty + i * 8][tx] = xb[(size_t)(c0 + ty + i * 8) * S_DIM + s0 + tx];
    __syncthreads();
#pragma unroll
    for (int i = 0; i < 4; ++i)
        tb[(size_t)(s0 + ty + i * 8) * C_DIM + c0 + tx] = (bf16t)tile[tx][ty + i * 8];
}

// ---------------------------------------------------------------------------
// GEMM: Y[M=S,N=C] = A[M,K=C] @ W[N,K]^T + bias.
//   Block 256 thr (8 waves), tile 128x128, K-step 32, double-buffered LDS
//   staged with async global->LDS DMA. Each wave: 64x32 tile, 8 WMMAs/K-step.
//   MODE 0: store bf16 [B,M,N];  MODE 1: store fp32 transposed [B,N,M].
// ---------------------------------------------------------------------------
template <int MODE>
__global__ void __launch_bounds__(256)
gemm_bf16_nt(const bf16t* __restrict__ A, const bf16t* __restrict__ W,
             const float* __restrict__ bias, bf16t* __restrict__ Ybf,
             float* __restrict__ Yt) {
    const int M = S_DIM, N = C_DIM, K = C_DIM;
    const int b  = blockIdx.z;
    const bf16t* Ab = A + (size_t)b * M * K;
    const int m0 = blockIdx.x * 128;
    const int n0 = blockIdx.y * 128;

    __shared__ __align__(16) bf16t As[2][128][40];   // +8 pad, 80B rows (16B aligned)
    __shared__ __align__(16) bf16t Ws[2][128][40];

    const int tid    = threadIdx.x;
    const int lane   = tid & 31;
    const int wid    = tid >> 5;
    const int wm     = wid >> 2;      // 0..1  (64-row slab)
    const int wn     = wid & 3;       // 0..3  (32-col slab)
    const int lanelo = lane & 15;
    const int half   = lane >> 4;

    const int lr = tid >> 1;          // 0..127: tile row this thread stages
    const int lc = (tid & 1) * 16;    // 0 or 16: col half

    const bf16t* gA = &Ab[(size_t)(m0 + lr) * K + lc];
    const bf16t* gW = &W [(size_t)(n0 + lr) * K + lc];

    // Prologue: stage tile 0 into buffer 0.
    copy16_g2l(&As[0][lr][lc],     gA);
    copy16_g2l(&As[0][lr][lc + 8], gA + 8);
    copy16_g2l(&Ws[0][lr][lc],     gW);
    copy16_g2l(&Ws[0][lr][lc + 8], gW + 8);

    v8f acc[4][2];
#pragma unroll
    for (int i = 0; i < 4; ++i)
#pragma unroll
        for (int j = 0; j < 2; ++j) acc[i][j] = v8f_zero();

    const int NIT = K / 32;
    for (int it = 0; it < NIT; ++it) {
        wait_async0();            // this wave's DMA for tile `it` complete
        __syncthreads();          // all waves' tile `it` in LDS; prev compute done

        if (it + 1 < NIT) {       // kick off next tile while computing this one
            const int nb = (it + 1) & 1;
            const bf16t* gA2 = gA + (it + 1) * 32;
            const bf16t* gW2 = gW + (it + 1) * 32;
            copy16_g2l(&As[nb][lr][lc],     gA2);
            copy16_g2l(&As[nb][lr][lc + 8], gA2 + 8);
            copy16_g2l(&Ws[nb][lr][lc],     gW2);
            copy16_g2l(&Ws[nb][lr][lc + 8], gW2 + 8);
        }

        const int cb = it & 1;
        v16bf afr[4], bfr[2];
#pragma unroll
        for (int i = 0; i < 4; ++i) {   // A 16x32: [k+half*8 .. +7], [k+16+half*8 ..]
            const bf16t* rp = &As[cb][wm * 64 + i * 16 + lanelo][half * 8];
            afr[i] = ld_frag16(rp, rp + 16);
        }
#pragma unroll
        for (int j = 0; j < 2; ++j) {   // B 32x16: [k+half*8 .. +7], [k+16+half*8 ..]
            const bf16t* rp = &Ws[cb][wn * 32 + j * 16 + lanelo][half * 8];
            bfr[j] = ld_frag16(rp, rp + 16);
        }
#pragma unroll
        for (int i = 0; i < 4; ++i)
#pragma unroll
            for (int j = 0; j < 2; ++j)
                acc[i][j] = __builtin_amdgcn_wmma_f32_16x16x32_bf16(
                    false, afr[i], false, bfr[j], (short)0, acc[i][j], false, false);
    }

#pragma unroll
    for (int i = 0; i < 4; ++i) {
#pragma unroll
        for (int j = 0; j < 2; ++j) {
            const int col = n0 + wn * 32 + j * 16 + lanelo;
            const float bv = bias[col];
#pragma unroll
            for (int r = 0; r < 8; ++r) {
                const int row = m0 + wm * 64 + i * 16 + r + half * 8;
                const float v = acc[i][j][r] + bv;
                if (MODE == 0)
                    Ybf[(size_t)b * M * N + (size_t)row * N + col] = (bf16t)v;
                else
                    Yt[(size_t)b * N * M + (size_t)col * M + row] = v;  // [B,C,S]
            }
        }
    }
}

// ---------------------------------------------------------------------------
// Flash attention. Block = (b, h, 128 Q rows), 256 thr = 8 waves x 16 rows.
// K tiles async-DMA'd once per block into double-buffered LDS (was: 8x
// redundant per-wave global reads). V staged transposed in LDS; P routed via
// per-wave LDS (C-layout -> A-layout). Q fragments persistent in registers.
// ---------------------------------------------------------------------------
__global__ void __launch_bounds__(256)
attention_kernel(const bf16t* __restrict__ Q, const bf16t* __restrict__ Km,
                 const bf16t* __restrict__ V, bf16t* __restrict__ AO) {
    const int b  = blockIdx.z;
    const int h  = blockIdx.y;
    const int q0 = blockIdx.x * 128;
    const int tid    = threadIdx.x;
    const int lane   = tid & 31;
    const int w      = tid >> 5;      // wave 0..7
    const int lanelo = lane & 15;
    const int half   = lane >> 4;

    const size_t bo = (size_t)b * S_DIM * C_DIM;
    const bf16t* Qb = Q  + bo;
    const bf16t* Kb = Km + bo;
    const bf16t* Vb = V  + bo;
    const int hc = h * HD;

    __shared__ __align__(16) bf16t Ks[2][64][72];    // K tile (row-major, dbl buf)
    __shared__ __align__(16) bf16t Vt[64][72];       // V tile transposed: [d][key]
    __shared__ __align__(16) bf16t Pt[8][16][72];    // per-wave P tile: [row][key]

    // Persistent Q A-fragments (16 rows x d=64 -> 2 fragments)
    v16bf aq[2];
    {
        const bf16t* qr = &Qb[(size_t)(q0 + w * 16 + lanelo) * C_DIM + hc];
#pragma unroll
        for (int kc = 0; kc < 2; ++kc) {
            const bf16t* p = qr + kc * 32 + half * 8;
            aq[kc] = ld_frag16(p, p + 16);
        }
    }

    v8f o[4];
    float mrow[8], lrow[8];
#pragma unroll
    for (int j = 0; j < 4; ++j) o[j] = v8f_zero();
#pragma unroll
    for (int r = 0; r < 8; ++r) { mrow[r] = -1e30f; lrow[r] = 0.f; }

    const float scale = 0.125f;          // 1/sqrt(64)
    const int skey = tid >> 2;           // 0..63: tile row this thread stages
    const int sd   = (tid & 3) * 16;     // d-chunk (16 elems)

    // Prologue: async-stage K tile 0 into buffer 0.
    {
        const bf16t* kp = &Kb[(size_t)skey * C_DIM + hc + sd];
        copy16_g2l(&Ks[0][skey][sd],     kp);
        copy16_g2l(&Ks[0][skey][sd + 8], kp + 8);
    }

    const int NT = S_DIM / 64;
    for (int it = 0; it < NT; ++it) {
        const int t0 = it * 64;

        // ---- stage V tile into registers (coalesced row reads) ----
        uint4 vr[2];
        const bf16t* vp = &Vb[(size_t)(t0 + skey) * C_DIM + hc + sd];
        vr[0] = *reinterpret_cast<const uint4*>(vp);
        vr[1] = *reinterpret_cast<const uint4*>(vp + 8);
        if (it + 1 < NT)                 // prefetch next V tile into caches
            __builtin_prefetch(&Vb[(size_t)(t0 + 64 + skey) * C_DIM + hc + sd], 0, 1);

        wait_async0();                   // this wave's K-tile DMA complete
        __syncthreads();                 // K tile ready; prior iter LDS reads done

        if (it + 1 < NT) {               // DMA next K tile during compute
            const int nb = (it + 1) & 1;
            const bf16t* kp = &Kb[(size_t)(t0 + 64 + skey) * C_DIM + hc + sd];
            copy16_g2l(&Ks[nb][skey][sd],     kp);
            copy16_g2l(&Ks[nb][skey][sd + 8], kp + 8);
        }

#pragma unroll
        for (int i = 0; i < 2; ++i) {    // transpose V into LDS: Vt[d][key]
            union { uint4 q; bf16t e[8]; } u; u.q = vr[i];
#pragma unroll
            for (int e2 = 0; e2 < 8; ++e2)
                Vt[sd + i * 8 + e2][skey] = u.e[e2];
        }

        // ---- S = Q @ K^T (B-frags from LDS K tile: contiguous d elems) ----
        const int cb = it & 1;
        v8f s[4];
#pragma unroll
        for (int j = 0; j < 4; ++j) s[j] = v8f_zero();
#pragma unroll
        for (int j = 0; j < 4; ++j) {
#pragma unroll
            for (int kc = 0; kc < 2; ++kc) {
                const bf16t* p = &Ks[cb][j * 16 + lanelo][kc * 32 + half * 8];
                v16bf bk = ld_frag16(p, p + 16);
                s[j] = __builtin_amdgcn_wmma_f32_16x16x32_bf16(
                    false, aq[kc], false, bk, (short)0, s[j], false, false);
            }
        }

        // ---- online softmax (C-layout: element r <-> row r + 8*half) ----
        float tmax[8];
#pragma unroll
        for (int r = 0; r < 8; ++r) {
            float mv = -1e30f;
#pragma unroll
            for (int j = 0; j < 4; ++j) { s[j][r] *= scale; mv = fmaxf(mv, s[j][r]); }
            tmax[r] = mv;
        }
#pragma unroll
        for (int off = 1; off < 16; off <<= 1)
#pragma unroll
            for (int r = 0; r < 8; ++r)
                tmax[r] = fmaxf(tmax[r], __shfl_xor(tmax[r], off, 32));

        float alpha[8];
#pragma unroll
        for (int r = 0; r < 8; ++r) {
            const float mn = fmaxf(mrow[r], tmax[r]);
            alpha[r] = __expf(mrow[r] - mn);
            mrow[r]  = mn;
        }

        float psum[8];
#pragma unroll
        for (int r = 0; r < 8; ++r) psum[r] = 0.f;
#pragma unroll
        for (int j = 0; j < 4; ++j)
#pragma unroll
            for (int r = 0; r < 8; ++r) {
                const float p = __expf(s[j][r] - mrow[r]);
                psum[r] += p;
                Pt[w][r + half * 8][j * 16 + lanelo] = (bf16t)p;
            }
#pragma unroll
        for (int off = 1; off < 16; off <<= 1)
#pragma unroll
            for (int r = 0; r < 8; ++r)
                psum[r] += __shfl_xor(psum[r], off, 32);
#pragma unroll
        for (int r = 0; r < 8; ++r) lrow[r] = lrow[r] * alpha[r] + psum[r];
#pragma unroll
        for (int j = 0; j < 4; ++j)
#pragma unroll
            for (int r = 0; r < 8; ++r) o[j][r] *= alpha[r];

        __syncthreads();                 // Vt + Pt visible / ordered

        // ---- O += P @ V ----
        v16bf ap[2];
#pragma unroll
        for (int kc = 0; kc < 2; ++kc) {
            const bf16t* p = &Pt[w][lanelo][kc * 32 + half * 8];
            ap[kc] = ld_frag16(p, p + 16);
        }
#pragma unroll
        for (int j = 0; j < 4; ++j)
#pragma unroll
            for (int kc = 0; kc < 2; ++kc) {
                const bf16t* p = &Vt[j * 16 + lanelo][kc * 32 + half * 8];
                v16bf bv = ld_frag16(p, p + 16);
                o[j] = __builtin_amdgcn_wmma_f32_16x16x32_bf16(
                    false, ap[kc], false, bv, (short)0, o[j], false, false);
            }
    }

    // ---- finalize: divide by l, store merged-heads bf16 [B,S,C] ----
#pragma unroll
    for (int j = 0; j < 4; ++j)
#pragma unroll
        for (int r = 0; r < 8; ++r) {
            const int row = q0 + w * 16 + r + half * 8;
            const int col = hc + j * 16 + lanelo;
            AO[bo + (size_t)row * C_DIM + col] = (bf16t)(o[j][r] / lrow[r]);
        }
}

// ---------------------------------------------------------------------------
extern "C" void kernel_launch(void* const* d_in, const int* in_sizes, int n_in,
                              void* d_out, int out_size, void* d_ws, size_t ws_size,
                              hipStream_t stream) {
    (void)in_sizes; (void)n_in; (void)out_size; (void)ws_size;
    const float* x  = (const float*)d_in[0];
    const float* Wq = (const float*)d_in[1];
    const float* bq = (const float*)d_in[2];
    const float* Wk = (const float*)d_in[3];
    const float* bk = (const float*)d_in[4];
    const float* Wv = (const float*)d_in[5];
    const float* bv = (const float*)d_in[6];
    const float* Wp = (const float*)d_in[7];
    const float* bp = (const float*)d_in[8];
    float* out = (float*)d_out;

    char* ws = (char*)d_ws;
    const size_t tokB = (size_t)BATCH * S_DIM * C_DIM * sizeof(bf16t);  // 8 MB
    const size_t wB   = (size_t)C_DIM * C_DIM * sizeof(bf16t);          // 512 KB
    bf16t* tok = (bf16t*)(ws);                       // reused as AO after QKV
    bf16t* Wqb = (bf16t*)(ws + tokB);
    bf16t* Wkb = (bf16t*)(ws + tokB + 1 * wB);
    bf16t* Wvb = (bf16t*)(ws + tokB + 2 * wB);
    bf16t* Wpb = (bf16t*)(ws + tokB + 3 * wB);
    bf16t* Qb  = (bf16t*)(ws + tokB + 4 * wB);
    bf16t* Kb  = (bf16t*)(ws + tokB + 4 * wB + 1 * tokB);
    bf16t* Vb  = (bf16t*)(ws + tokB + 4 * wB + 2 * tokB);
    bf16t* AOb = tok;   // tok dead after V projection -> alias (total ws ~34 MB)

    cvt_w<<<dim3((C_DIM * C_DIM + 255) / 256), 256, 0, stream>>>(
        Wq, Wk, Wv, Wp, Wqb, Wkb, Wvb, Wpb);
    tok_transpose<<<dim3(S_DIM / 32, C_DIM / 32, BATCH), dim3(32, 8), 0, stream>>>(x, tok);

    dim3 ggrid(S_DIM / 128, C_DIM / 128, BATCH);
    gemm_bf16_nt<0><<<ggrid, 256, 0, stream>>>(tok, Wqb, bq, Qb, nullptr);
    gemm_bf16_nt<0><<<ggrid, 256, 0, stream>>>(tok, Wkb, bk, Kb, nullptr);
    gemm_bf16_nt<0><<<ggrid, 256, 0, stream>>>(tok, Wvb, bv, Vb, nullptr);

    attention_kernel<<<dim3(S_DIM / 128, NH, BATCH), 256, 0, stream>>>(Qb, Kb, Vb, AOb);

    gemm_bf16_nt<1><<<ggrid, 256, 0, stream>>>(AOb, Wpb, bp, nullptr, out);
}
